// VectorQuantizer_86277303042185
// MI455X (gfx1250) — matrix operand order repaired and checked by reading, not compiled, not run on hardware
//
#include <hip/hip_runtime.h>
#include <hip/hip_bf16.h>
#include <math.h>

typedef __attribute__((ext_vector_type(16))) _Float16 v16h;
typedef __attribute__((ext_vector_type(8)))  float    v8f;

#define VQ_B 8
#define VQ_D 256
#define VQ_T 2048
#define VQ_K 8192
#define VQ_N (VQ_B * VQ_T)      // 16384 tokens
#define NTILES (VQ_N / 16)      // 1024 token tiles
#define CTILES (VQ_K / 16)      // 512 code tiles
#define NSTAGE (VQ_K / 32)      // 256 double-tile stages
#define KCH (VQ_D / 32)         // 8 k-chunks of 32 per WMMA
#define CB_SCALE 4096.0f        // lift tiny codebook values into healthy f16 range
#define INV_CB_SCALE (1.0f / 4096.0f)

// Use the CDNA5 async copy engine (global_load_async_to_lds_b128 + ASYNCcnt)
#define VQ_ASYNC 1

// ---------------------------------------------------------------- zero scratch
__global__ void vq_zero(int* counts, float* sumsq) {
    int i = blockIdx.x * blockDim.x + threadIdx.x;
    if (i < VQ_K) counts[i] = 0;
    if (i == 0) *sumsq = 0.0f;
}

// ----------------------------------------------------- codebook row norms f32
__global__ __launch_bounds__(256) void vq_cnorm(const float* __restrict__ cb,
                                                float* __restrict__ cnorms) {
    const int lane = threadIdx.x & 31;
    const int wave = threadIdx.x >> 5;
    const int k = blockIdx.x * 8 + wave;
    const float* row = cb + (size_t)k * VQ_D;
    float s = 0.0f;
    for (int j = lane; j < VQ_D; j += 32) { float v = row[j]; s = fmaf(v, v, s); }
    for (int off = 16; off; off >>= 1) s += __shfl_xor(s, off);
    if (lane == 0) cnorms[k] = s;
}

// ------------------------- pack x[B,D,T] -> f16 A fragments (ISA 16x32 layout)
// Apack[((tt*8 + kc)*32 + lane)*16 + h] ; lane holds its 8 A VGPRs contiguously
__global__ __launch_bounds__(256) void vq_pack_x(const float* __restrict__ x,
                                                 _Float16* __restrict__ Apack) {
    int idx = blockIdx.x * 256 + threadIdx.x;          // 0 .. N*D-1
    int h  = idx & 15;
    int l  = (idx >> 4) & 31;
    int kc = (idx >> 9) & 7;
    int tt = idx >> 12;
    int v = h >> 1, s = h & 1;
    int M, Kl;
    if (l < 16) { M = l;      Kl = (v < 4) ? (2 * v + s)      : (16 + 2 * (v - 4) + s); }
    else        { M = l - 16; Kl = (v < 4) ? (8 + 2 * v + s)  : (24 + 2 * (v - 4) + s); }
    int d = kc * 32 + Kl;
    int n = tt * 16 + M;
    int b = n >> 11;               // T = 2048
    int t = n & (VQ_T - 1);
    Apack[idx] = (_Float16)x[((size_t)b * VQ_D + d) * VQ_T + t];
}

// --------------- pack codebook -> scaled f16 B fragments (ISA 16x32 layout)
__global__ __launch_bounds__(256) void vq_pack_c(const float* __restrict__ cb,
                                                 _Float16* __restrict__ Bpack) {
    int idx = blockIdx.x * 256 + threadIdx.x;          // 0 .. K*D-1
    int h  = idx & 15;
    int l  = (idx >> 4) & 31;
    int kc = (idx >> 9) & 7;
    int ct = idx >> 12;
    int col = l & 15;
    int Kl  = (h < 8) ? (8 * (l >> 4) + h) : (16 + 8 * (l >> 4) + (h - 8));   // lanes16-31 -> K+8
    int d = kc * 32 + Kl;
    int k = ct * 16 + col;
    Bpack[idx] = (_Float16)(cb[(size_t)k * VQ_D + d] * CB_SCALE);
}

#if VQ_ASYNC
// Issue one 16 KB stage copy (two 16-code fragment tiles) via the async engine.
// Each of the 256 threads moves 64 contiguous bytes; the immediate offset adds
// to BOTH the global and LDS address (ISA §15.18 async pseudocode), which
// matches our identical contiguous layouts on both sides.
__device__ __forceinline__ void vq_stage_async(const uint4* __restrict__ seg,
                                               uint32_t lds_byte, int tid) {
    uint64_t g = (uint64_t)(const void*)(seg + tid * 4);
    uint32_t l = lds_byte + (uint32_t)tid * 64u;
    asm volatile("global_load_async_to_lds_b128 %0, %1, off"
                 :: "v"(l), "v"(g) : "memory");
    asm volatile("global_load_async_to_lds_b128 %0, %1, off offset:16"
                 :: "v"(l), "v"(g) : "memory");
    asm volatile("global_load_async_to_lds_b128 %0, %1, off offset:32"
                 :: "v"(l), "v"(g) : "memory");
    asm volatile("global_load_async_to_lds_b128 %0, %1, off offset:48"
                 :: "v"(l), "v"(g) : "memory");
}
#endif

// ----------------------- fused distance-GEMM + argmin (the WMMA heavy kernel)
// 8 waves/block, 16 tokens per wave held as A fragments in registers for the
// whole K loop. Each stage covers 32 codes: two independent WMMA accumulation
// chains keep the XDL pipe busy; LDS is double-buffered and filled by the
// async copy engine one stage ahead (single barrier per stage).
__global__ __launch_bounds__(256) void vq_argmin(const _Float16* __restrict__ Apack,
                                                 const _Float16* __restrict__ Bpack,
                                                 const float* __restrict__ cnorms,
                                                 int* __restrict__ codes,
                                                 float* __restrict__ codes_f) {
    __shared__ __align__(16) _Float16 lds[2][2 * KCH * 32 * 16];   // 2 x 16 KB
    const int tid  = threadIdx.x;
    const int lane = tid & 31;
    const int wave = tid >> 5;
    const int tt   = blockIdx.x * 8 + wave;                 // 16-token tile/wave

    const v16h* ap = (const v16h*)Apack + (size_t)tt * (KCH * 32);
    v16h a[KCH];
#pragma unroll
    for (int kc = 0; kc < KCH; ++kc) a[kc] = ap[kc * 32 + lane];

    float best[8];
    int   bidx[8];
#pragma unroll
    for (int m = 0; m < 8; ++m) { best[m] = 3.4e38f; bidx[m] = 0; }

    const uint4* bsrc = (const uint4*)Bpack;
    const int col = lane & 15;

#if VQ_ASYNC
    const uint32_t lbase[2] = { (uint32_t)(size_t)&lds[0][0],
                                (uint32_t)(size_t)&lds[1][0] };
    vq_stage_async(bsrc, lbase[0], tid);                    // prime stage 0
#endif

    for (int st = 0; st < NSTAGE; ++st) {
        const int cur = st & 1;
#if VQ_ASYNC
        // our own stage-(st) copies are the only outstanding async ops
        asm volatile("s_wait_asynccnt 0x0" ::: "memory");
        __syncthreads();                                    // all waves' data in
        if (st + 1 < NSTAGE)                                // prefetch next stage
            vq_stage_async(bsrc + (size_t)(st + 1) * 1024, lbase[(st + 1) & 1], tid);
#else
        {
            const uint4* seg = bsrc + (size_t)st * 1024;
            uint4* ldsv = (uint4*)&lds[cur][0];
#pragma unroll
            for (int j = 0; j < 4; ++j) ldsv[tid * 4 + j] = seg[tid * 4 + j];
        }
        __syncthreads();
#endif
        const v16h* lf = (const v16h*)&lds[cur][0];
        v8f acc0 = {};
        v8f acc1 = {};
#pragma unroll
        for (int kc = 0; kc < KCH; ++kc) {
            v16h bf0 = lf[kc * 32 + lane];          // code tile 2*st
            v16h bf1 = lf[256 + kc * 32 + lane];    // code tile 2*st+1
            acc0 = __builtin_amdgcn_wmma_f32_16x16x32_f16(
                false, a[kc], false, bf0, (short)0, acc0, false, false);
            acc1 = __builtin_amdgcn_wmma_f32_16x16x32_f16(
                false, a[kc], false, bf1, (short)0, acc1, false, false);
        }

        // score = ||c||^2 - 2 * (x . c)   (||x||^2 constant -> dropped)
        const float cn0 = cnorms[st * 32 + col];
        const float cn1 = cnorms[st * 32 + 16 + col];
        const int code0 = st * 32 + col;
        const int code1 = st * 32 + 16 + col;
#pragma unroll
        for (int m = 0; m < 8; ++m) {
            float s0 = fmaf(-2.0f * INV_CB_SCALE, acc0[m], cn0);
            float s1 = fmaf(-2.0f * INV_CB_SCALE, acc1[m], cn1);
            if (s0 < best[m]) { best[m] = s0; bidx[m] = code0; }
            if (s1 < best[m]) { best[m] = s1; bidx[m] = code1; }
        }
#if !VQ_ASYNC
        __syncthreads();
#endif
    }

    // cross-lane argmin: each lane's 8 values share one code column, so reduce
    // over the 16 lanes of each half (masks <=8 never cross the 16-lane split)
#pragma unroll
    for (int m = 0; m < 8; ++m) {
        for (int off = 8; off; off >>= 1) {
            float od = __shfl_xor(best[m], off);
            int   oi = __shfl_xor(bidx[m], off);
            if (od < best[m] || (od == best[m] && oi < bidx[m])) {
                best[m] = od; bidx[m] = oi;
            }
        }
    }
    if (lane == 0) {
#pragma unroll
        for (int m = 0; m < 8; ++m) {
            int n = tt * 16 + m;
            codes[n] = bidx[m]; codes_f[n] = (float)bidx[m];
        }
    } else if (lane == 16) {
#pragma unroll
        for (int m = 0; m < 8; ++m) {
            int n = tt * 16 + 8 + m;
            codes[n] = bidx[m]; codes_f[n] = (float)bidx[m];
        }
    }
}

// ---------------------- gather quantized vectors + fused MSE partial reduction
__global__ __launch_bounds__(256) void vq_gather(const float* __restrict__ x,
                                                 const float* __restrict__ cb,
                                                 const int* __restrict__ codes,
                                                 float* __restrict__ outq,
                                                 float* __restrict__ sumsq) {
    __shared__ float red[8];
    int gid = blockIdx.x * 256 + threadIdx.x;      // (b,d,t) flat, t fastest
    int t = gid & (VQ_T - 1);
    int d = (gid >> 11) & (VQ_D - 1);
    int b = gid >> 19;                             // D*T = 2^19
    int n = b * VQ_T + t;
    int k = codes[n];
    float q  = cb[(size_t)k * VQ_D + d];
    float xv = x[gid];
    outq[gid] = q;                                 // straight-through forward == q
    float df = q - xv;
    float sq = df * df;
    for (int off = 16; off; off >>= 1) sq += __shfl_xor(sq, off);
    int lane = threadIdx.x & 31, wave = threadIdx.x >> 5;
    if (lane == 0) red[wave] = sq;
    __syncthreads();
    if (threadIdx.x == 0) {
        float s = 0.0f;
        for (int w = 0; w < 8; ++w) s += red[w];
        atomicAdd(sumsq, s);
    }
}

// -------------------------------------------------------------- code histogram
__global__ void vq_count(const int* __restrict__ codes, int* __restrict__ counts) {
    int n = blockIdx.x * blockDim.x + threadIdx.x;
    if (n < VQ_N) atomicAdd(&counts[codes[n]], 1);
}

// ------------------------------------------- losses + perplexity finalization
__global__ __launch_bounds__(256) void vq_final(const int* __restrict__ counts,
                                                const float* __restrict__ sumsq,
                                                float* __restrict__ out3) {
    __shared__ float red[8];
    float s = 0.0f;
    for (int k = threadIdx.x; k < VQ_K; k += 256) {
        float p = (float)counts[k] / (float)VQ_N;
        s += p * logf(p + 1e-10f);
    }
    for (int off = 16; off; off >>= 1) s += __shfl_xor(s, off);
    if ((threadIdx.x & 31) == 0) red[threadIdx.x >> 5] = s;
    __syncthreads();
    if (threadIdx.x == 0) {
        float e = 0.0f;
        for (int w = 0; w < 8; ++w) e += red[w];
        float loss = *sumsq / (float)(VQ_B * VQ_D * VQ_T);
        out3[0] = loss;          // commitment loss (== codebook loss forward)
        out3[1] = loss;          // codebook loss
        out3[2] = expf(-e);      // perplexity
    }
}

extern "C" void kernel_launch(void* const* d_in, const int* in_sizes, int n_in,
                              void* d_out, int out_size, void* d_ws, size_t ws_size,
                              hipStream_t stream) {
    const float* x  = (const float*)d_in[0];   // [B, D, T]
    const float* cb = (const float*)d_in[1];   // [K, D]
    float* out = (float*)d_out;
    char*  ws  = (char*)d_ws;

    // workspace layout (~12.2 MB total)
    _Float16* Apack  = (_Float16*)(ws);                     // N*D halves  = 8 MB
    _Float16* Bpack  = (_Float16*)(ws + 8388608);           // K*D halves  = 4 MB
    float*    cnorms = (float*)   (ws + 12582912);          // K f32       = 32 KB
    int*      codes  = (int*)     (ws + 12615680);          // N i32       = 64 KB
    int*      counts = (int*)     (ws + 12681216);          // K i32       = 32 KB
    float*    sumsq  = (float*)   (ws + 12713984);          // 1 f32

    float* outq     = out;                                  // [B, D, T]
    float* outcodes = out + (size_t)VQ_B * VQ_D * VQ_T;     // [B, T] as f32
    float* out3     = outcodes + VQ_N;                      // 3 scalars

    vq_zero  <<<(VQ_K + 255) / 256, 256, 0, stream>>>(counts, sumsq);
    vq_cnorm <<<VQ_K / 8,           256, 0, stream>>>(cb, cnorms);
    vq_pack_x<<<(VQ_N * VQ_D) / 256, 256, 0, stream>>>(x, Apack);
    vq_pack_c<<<(VQ_K * VQ_D) / 256, 256, 0, stream>>>(cb, Bpack);
    vq_argmin<<<NTILES / 8,          256, 0, stream>>>(Apack, Bpack, cnorms, codes, outcodes);
    vq_gather<<<(VQ_B * VQ_D * VQ_T) / 256, 256, 0, stream>>>(x, cb, codes, outq, sumsq);
    vq_count <<<VQ_N / 256,          256, 0, stream>>>(codes, counts);
    vq_final <<<1,                   256, 0, stream>>>(counts, sumsq, out3);
}
